// MultiHeadAttention_6786048328299
// MI455X (gfx1250) — matrix operand order repaired and checked by reading, not compiled, not run on hardware
//
#include <hip/hip_runtime.h>
#include <math.h>

// MultiHeadAttention forward for MI455X (gfx1250, wave32, WMMA).
//   B=4, S=2048, EMB=1024, H=16, Dh=64.
// Pipeline: 3x projection GEMM (f32 in -> bf16 [B,H,S,D] scratch, WMMA bf16)
//           1x fused flash-attention (WMMA scores + online softmax + WMMA PV)
//           1x output GEMM (bf16 in -> f32 d_out + bias)
// bf16 tiles that need no conversion are moved global->LDS with the CDNA5
// async-load path (ASYNCcnt), converting tiles go through VGPRs.

typedef __bf16 v16bf __attribute__((ext_vector_type(16)));
typedef __bf16 v8bf  __attribute__((ext_vector_type(8)));
typedef float  v8f   __attribute__((ext_vector_type(8)));
typedef int    i4v   __attribute__((vector_size(16)));   // int4, 16 bytes

#define EMB  1024
#define NB   4
#define SEQ  2048
#define NH   16
#define DH   64
#define MTOT (NB * SEQ)   // 8192 rows in all [M,1024] GEMMs

#define AS_GLOBAL __attribute__((address_space(1)))
#define AS_LOCAL  __attribute__((address_space(3)))

static __device__ __forceinline__ v16bf cat16(v8bf lo, v8bf hi) {
  return __builtin_shufflevector(lo, hi, 0,1,2,3,4,5,6,7,8,9,10,11,12,13,14,15);
}

static __device__ __forceinline__ v8f wmma_bf16(v16bf a, v16bf b, v8f c) {
  // D(16x16,f32) = A(16x32,bf16) * B(32x16,bf16) + C
  return __builtin_amdgcn_wmma_f32_16x16x32_bf16(
      /*neg_a=*/false, a, /*neg_b=*/false, b,
      /*c_mod=*/(short)0, c, /*reuse_a=*/false, /*reuse_b=*/false);
}

// ---- CDNA5 async global->LDS copy (16 bytes / lane), ASYNCcnt tracked ------
static __device__ __forceinline__ void gld_async_b128(void* lds_dst,
                                                      const void* gsrc) {
#if __has_builtin(__builtin_amdgcn_global_load_async_to_lds_b128)
  __builtin_amdgcn_global_load_async_to_lds_b128(
      (AS_GLOBAL i4v*)gsrc, (AS_LOCAL i4v*)lds_dst,
      /*offset=*/0, /*cpol=*/0);
#else
  const unsigned loff = (unsigned)(unsigned long long)(AS_LOCAL void*)lds_dst;
  asm volatile("global_load_async_to_lds_b128 %0, %1, off"
               :
               : "v"(loff), "v"(gsrc)
               : "memory");
#endif
}

static __device__ __forceinline__ void wait_async0() {
#if __has_builtin(__builtin_amdgcn_s_wait_asynccnt)
  __builtin_amdgcn_s_wait_asynccnt(0);
#else
  asm volatile("s_wait_asynccnt 0x0" ::: "memory");
#endif
}

// ---------------------------------------------------------------------------
// GEMM: out[M,N] = A[M,K] @ W[N,K]^T + bias   (torch Linear convention)
// Tile: 128x128 per workgroup, BK=32, 8 waves, each wave 64x32 (4x2 WMMA).
// A_BF16:     A already bf16 (attention output) vs f32 (raw inputs).
// SPLIT_HEADS: write bf16 to [B,H,S,Dh] scratch vs f32 to [M,N] d_out.
// ---------------------------------------------------------------------------
template <bool A_BF16, bool SPLIT_HEADS>
__global__ __launch_bounds__(256)
void gemm_wmma_kernel(const void* __restrict__ Aptr,
                      const float* __restrict__ W,
                      const float* __restrict__ bias,
                      void* __restrict__ outp) {
  __shared__ __bf16 sA[128][32];   // A tile  [m][k]
  __shared__ __bf16 sB[128][32];   // W tile  [n][k]  (== B^T storage)

  const int tid   = threadIdx.x;
  const int wave  = tid >> 5;
  const int lane  = tid & 31;
  const int half  = lane >> 4;     // 0: lanes 0-15, 1: lanes 16-31
  const int l15   = lane & 15;
  const int m_blk = blockIdx.y * 128;
  const int n_blk = blockIdx.x * 128;
  const int wm    = (wave >> 2) * 64;   // wave M offset: 0 / 64
  const int wn    = (wave & 3) * 32;    // wave N offset: 0/32/64/96

  v8f acc[4][2];
#pragma unroll
  for (int i = 0; i < 4; ++i)
#pragma unroll
    for (int j = 0; j < 2; ++j) acc[i][j] = {};

  const float*  Af = (const float*)Aptr;
  const __bf16* Ab = (const __bf16*)Aptr;

  for (int k0 = 0; k0 < EMB; k0 += 32) {
    // ---- cooperative A tile load ----
    if (A_BF16) {
      // bf16 source: async global->LDS, 16B per copy, no VGPR transit
#pragma unroll
      for (int r = 0; r < 2; ++r) {
        const int row = (tid >> 2) + 64 * r;
        const int seg = (tid & 3) * 8;
        gld_async_b128(&sA[row][seg],
                       Ab + (size_t)(m_blk + row) * EMB + k0 + seg);
      }
    } else {
#pragma unroll
      for (int r = 0; r < 4; ++r) {
        const int row = (tid >> 3) + 32 * r;
        const int c4  = (tid & 7) * 4;
        const float4 f = *reinterpret_cast<const float4*>(
            Af + (size_t)(m_blk + row) * EMB + k0 + c4);
        sA[row][c4 + 0] = (__bf16)f.x;
        sA[row][c4 + 1] = (__bf16)f.y;
        sA[row][c4 + 2] = (__bf16)f.z;
        sA[row][c4 + 3] = (__bf16)f.w;
      }
    }
    // ---- cooperative W tile load (f32 -> bf16 conversion, VGPR path) ----
#pragma unroll
    for (int r = 0; r < 4; ++r) {
      const int row = (tid >> 3) + 32 * r;
      const int c4  = (tid & 7) * 4;
      const float4 f = *reinterpret_cast<const float4*>(
          W + (size_t)(n_blk + row) * EMB + k0 + c4);
      sB[row][c4 + 0] = (__bf16)f.x;
      sB[row][c4 + 1] = (__bf16)f.y;
      sB[row][c4 + 2] = (__bf16)f.z;
      sB[row][c4 + 3] = (__bf16)f.w;
    }
    if (A_BF16) wait_async0();
    __syncthreads();

    // ---- build fragments (ISA 7.12.2 layouts) + 8 WMMA per wave ----
    v16bf afr[4];
#pragma unroll
    for (int ms = 0; ms < 4; ++ms) {
      const int m  = wm + ms * 16 + l15;
      const int kb = half * 8;                       // A: kbase 0 / 8
      v8bf lo = *reinterpret_cast<const v8bf*>(&sA[m][kb]);       // k=kb..kb+7
      v8bf hi = *reinterpret_cast<const v8bf*>(&sA[m][kb + 16]);  // k=kb+16..+23
      afr[ms] = cat16(lo, hi);
    }
    v16bf bfr[2];
#pragma unroll
    for (int ns = 0; ns < 2; ++ns) {
      const int n  = wn + ns * 16 + l15;
      const int kb = half * 8;                       // B: kbase 0 / 8
      v8bf lo = *reinterpret_cast<const v8bf*>(&sB[n][kb]);
      v8bf hi = *reinterpret_cast<const v8bf*>(&sB[n][kb + 16]);
      bfr[ns] = cat16(lo, hi);
    }
#pragma unroll
    for (int ms = 0; ms < 4; ++ms)
#pragma unroll
      for (int ns = 0; ns < 2; ++ns)
        acc[ms][ns] = wmma_bf16(afr[ms], bfr[ns], acc[ms][ns]);

    __syncthreads();
  }

  // ---- epilogue: bias add, write out (C/D layout: m = r + 8*half, n = l15) --
#pragma unroll
  for (int ms = 0; ms < 4; ++ms) {
#pragma unroll
    for (int ns = 0; ns < 2; ++ns) {
      const int gn = n_blk + wn + ns * 16 + l15;
      const float bv = bias[gn];
#pragma unroll
      for (int r = 0; r < 8; ++r) {
        const int gm = m_blk + wm + ms * 16 + half * 8 + r;
        const float valf = acc[ms][ns][r] + bv;
        if (SPLIT_HEADS) {
          const int b = gm >> 11, s = gm & 2047;
          const int h = gn >> 6,  d = gn & 63;
          ((__bf16*)outp)[((size_t)((b * NH + h) * SEQ + s)) * DH + d] =
              (__bf16)valf;
        } else {
          ((float*)outp)[(size_t)gm * EMB + gn] = valf;
        }
      }
    }
  }
}

// ---------------------------------------------------------------------------
// Flash attention: one workgroup per (b*h, 128-query-row tile).
// 8 waves, each owns 16 query rows; KV processed in 64-row tiles.
// scale = 1/sqrt(EMB) = 0.03125 (reference divides by sqrt(emb), not head_dim)
// ---------------------------------------------------------------------------
__global__ __launch_bounds__(256)
void attn_wmma_kernel(const __bf16* __restrict__ q,
                      const __bf16* __restrict__ k,
                      const __bf16* __restrict__ v,
                      __bf16* __restrict__ o) {
  __shared__ __bf16 sQ[128][64];       // 16 KB
  __shared__ __bf16 sK[64][64];        //  8 KB   [kvrow][d]
  __shared__ __bf16 sVt[64][72];       //  9 KB   transposed: [d][kvrow], +pad
  __shared__ __bf16 sP[8][16][64];     // 16 KB   per-wave prob staging

  const int tid  = threadIdx.x;
  const int wave = tid >> 5;
  const int lane = tid & 31;
  const int half = lane >> 4;
  const int l15  = lane & 15;

  const int qt = blockIdx.x;               // 0..15 query tile
  const int bh = blockIdx.y;               // 0..63 (b*16+h)
  const size_t base  = (size_t)bh * SEQ * DH;
  const int    qbase = qt * 128;

  // ---- Q tile: async global->LDS (bf16, no conversion needed) ----
#pragma unroll
  for (int r = 0; r < 4; ++r) {
    const int row = (tid >> 3) + 32 * r;
    const int seg = (tid & 7) * 8;
    gld_async_b128(&sQ[row][seg],
                   q + base + (size_t)(qbase + row) * DH + seg);
  }
  wait_async0();
  __syncthreads();

  v16bf qf[2];   // two d-halves of the 16x64 Q tile, kept in VGPRs all loop
#pragma unroll
  for (int dh = 0; dh < 2; ++dh) {
    const int qrow = wave * 16 + l15;
    const int off  = dh * 32 + half * 8;
    v8bf lo = *reinterpret_cast<const v8bf*>(&sQ[qrow][off]);
    v8bf hi = *reinterpret_cast<const v8bf*>(&sQ[qrow][off + 16]);
    qf[dh] = cat16(lo, hi);
  }

  float m_run[8], l_run[8];
#pragma unroll
  for (int r = 0; r < 8; ++r) { m_run[r] = -1e30f; l_run[r] = 0.0f; }
  v8f oacc[4];
#pragma unroll
  for (int d = 0; d < 4; ++d) oacc[d] = {};

  const float scale = 0.03125f;   // 1/sqrt(1024)

  for (int j = 0; j < SEQ / 64; ++j) {
    __syncthreads();   // previous iteration done reading sK/sVt
    // ---- K tile: async copy; V tile: VGPR transit (transpose) ----
#pragma unroll
    for (int r = 0; r < 2; ++r) {
      const int row = (tid >> 3) + 32 * r;
      const int seg = (tid & 7) * 8;
      gld_async_b128(&sK[row][seg],
                     k + base + (size_t)(j * 64 + row) * DH + seg);
      v8bf vv = *reinterpret_cast<const v8bf*>(
          v + base + (size_t)(j * 64 + row) * DH + seg);
#pragma unroll
      for (int e = 0; e < 8; ++e) sVt[seg + e][row] = vv[e];
    }
    if (j + 1 < SEQ / 64) {   // hint next KV tile into cache
      __builtin_prefetch(k + base + (size_t)((j + 1) * 64) * DH + tid * 16, 0, 1);
      __builtin_prefetch(v + base + (size_t)((j + 1) * 64) * DH + tid * 16, 0, 1);
    }
    wait_async0();
    __syncthreads();

    // ---- scores: 16(q) x 64(kv), contraction over d=64 (2 WMMA k-steps) ----
    v8f sc[4];
#pragma unroll
    for (int nt = 0; nt < 4; ++nt) {
      sc[nt] = {};
#pragma unroll
      for (int ds = 0; ds < 2; ++ds) {
        const int n   = nt * 16 + l15;      // key row within tile (B column)
        const int off = ds * 32 + half * 8;
        v8bf lo = *reinterpret_cast<const v8bf*>(&sK[n][off]);
        v8bf hi = *reinterpret_cast<const v8bf*>(&sK[n][off + 16]);
        sc[nt] = wmma_bf16(qf[ds], cat16(lo, hi), sc[nt]);
      }
      sc[nt] *= scale;
    }

    // ---- online softmax: rows live per C/D element, reduce over lanes ----
    float m_new[8], alpha[8];
#pragma unroll
    for (int r = 0; r < 8; ++r) {
      float a = fmaxf(fmaxf(sc[0][r], sc[1][r]), fmaxf(sc[2][r], sc[3][r]));
#pragma unroll
      for (int off = 8; off >= 1; off >>= 1)
        a = fmaxf(a, __shfl_xor(a, off, 32));   // stays within 16-lane group
      m_new[r] = fmaxf(m_run[r], a);
      alpha[r] = __expf(m_run[r] - m_new[r]);
      m_run[r] = m_new[r];
      l_run[r] *= alpha[r];
    }
#pragma unroll
    for (int dt = 0; dt < 4; ++dt)
#pragma unroll
      for (int r = 0; r < 8; ++r) oacc[dt][r] *= alpha[r];

    float rs[8];
#pragma unroll
    for (int r = 0; r < 8; ++r) rs[r] = 0.0f;
#pragma unroll
    for (int nt = 0; nt < 4; ++nt) {
#pragma unroll
      for (int r = 0; r < 8; ++r) {
        const float p = __expf(sc[nt][r] - m_new[r]);
        rs[r] += p;
        // stage P in A-readable layout (per-wave region, DS in-order)
        sP[wave][half * 8 + r][nt * 16 + l15] = (__bf16)p;
      }
    }
#pragma unroll
    for (int r = 0; r < 8; ++r) {
      float a = rs[r];
#pragma unroll
      for (int off = 8; off >= 1; off >>= 1) a += __shfl_xor(a, off, 32);
      l_run[r] += a;
    }

    // ---- PV: O[16x64] += P[16x64] @ V[64x64], 2 k-steps x 4 d-tiles ----
#pragma unroll
    for (int kh = 0; kh < 2; ++kh) {
      const int offp = kh * 32 + half * 8;
      v8bf plo = *reinterpret_cast<const v8bf*>(&sP[wave][l15][offp]);
      v8bf phi = *reinterpret_cast<const v8bf*>(&sP[wave][l15][offp + 16]);
      const v16bf pf = cat16(plo, phi);
#pragma unroll
      for (int dt = 0; dt < 4; ++dt) {
        const int d    = dt * 16 + l15;
        const int offv = kh * 32 + half * 8;
        v8bf vlo = *reinterpret_cast<const v8bf*>(&sVt[d][offv]);
        v8bf vhi = *reinterpret_cast<const v8bf*>(&sVt[d][offv + 16]);
        oacc[dt] = wmma_bf16(pf, cat16(vlo, vhi), oacc[dt]);
      }
    }
  }

  // ---- normalize and write back to [B,S,EMB] bf16 scratch ----
  const int b = bh >> 4, h = bh & 15;
#pragma unroll
  for (int dt = 0; dt < 4; ++dt) {
#pragma unroll
    for (int r = 0; r < 8; ++r) {
      const int srow = qbase + wave * 16 + half * 8 + r;
      const int col  = h * 64 + dt * 16 + l15;
      const float valf = oacc[dt][r] / l_run[r];
      o[(size_t)(b * SEQ + srow) * EMB + col] = (__bf16)valf;
    }
  }
}

// ---------------------------------------------------------------------------
extern "C" void kernel_launch(void* const* d_in, const int* in_sizes, int n_in,
                              void* d_out, int out_size, void* d_ws, size_t ws_size,
                              hipStream_t stream) {
  const float* query = (const float*)d_in[0];
  const float* keyi  = (const float*)d_in[1];
  const float* vali  = (const float*)d_in[2];
  const float* Wq    = (const float*)d_in[3];
  const float* bq    = (const float*)d_in[4];
  const float* Wk    = (const float*)d_in[5];
  const float* bk    = (const float*)d_in[6];
  const float* Wv    = (const float*)d_in[7];
  const float* bv    = (const float*)d_in[8];
  const float* Wo    = (const float*)d_in[9];
  const float* bo    = (const float*)d_in[10];

  char* ws = (char*)d_ws;
  const size_t qkvBytes = (size_t)MTOT * EMB * sizeof(__bf16);  // 16 MB each
  __bf16* qs   = (__bf16*)(ws);
  __bf16* ks   = (__bf16*)(ws + qkvBytes);
  __bf16* vs   = (__bf16*)(ws + 2 * qkvBytes);
  __bf16* attn = (__bf16*)(ws + 3 * qkvBytes);

  const dim3 blk(256);
  const dim3 gGrid(EMB / 128, MTOT / 128);   // (8, 64)

  gemm_wmma_kernel<false, true><<<gGrid, blk, 0, stream>>>(query, Wq, bq, qs);
  gemm_wmma_kernel<false, true><<<gGrid, blk, 0, stream>>>(keyi,  Wk, bk, ks);
  gemm_wmma_kernel<false, true><<<gGrid, blk, 0, stream>>>(vali,  Wv, bv, vs);

  const dim3 aGrid(SEQ / 128, NB * NH);      // (16, 64)
  attn_wmma_kernel<<<aGrid, blk, 0, stream>>>(qs, ks, vs, attn);

  gemm_wmma_kernel<true, false><<<gGrid, blk, 0, stream>>>(attn, Wo, bo, d_out);
}
